// MSCMADeformBlock_61950608277724
// MI455X (gfx1250) — matrix operand order repaired and checked by reading, not compiled, not run on hardware
//
#include <hip/hip_runtime.h>
#include <cmath>

#define DEV __device__ __forceinline__

typedef __attribute__((ext_vector_type(16))) __bf16 v16bf;
typedef __attribute__((ext_vector_type(8)))  float  v8f;

// ---------------------------------------------------------------- constants
static constexpr int  Bsz   = 8;
static constexpr int  Lt    = 32;
static constexpr int  Hh    = 8;
static constexpr long NVv   = 9072;           // 72*96 + 36*48 + 18*24
static constexpr long BN    = Bsz * NVv;      // 72576 rows
static constexpr long CHUNK = 24192;          // BN / 3, divisible by 128

// ---------------------------------------------------------------- helpers
DEV unsigned short f2bf(float f) {
  unsigned int u = __float_as_uint(f);
  u += 0x7FFFu + ((u >> 16) & 1u);            // round-to-nearest-even
  return (unsigned short)(u >> 16);
}

union ABFrag { v16bf v; uint4 q[2]; };

// A fragment: 16x32 bf16 tile at (row0, k0) of row-major [*, ldk] matrix.
// Lane m (<16): row m, K {0..7, 16..23}; lane m+16: row m, K {8..15, 24..31}.
DEV v16bf load_frag_a(const unsigned short* base, int ldk, int row0, int k0, int lane) {
  ABFrag f;
  const unsigned short* p = base + (long)(row0 + (lane & 15)) * ldk + k0 + ((lane >> 4) << 3);
  f.q[0] = *reinterpret_cast<const uint4*>(p);
  f.q[1] = *reinterpret_cast<const uint4*>(p + 16);
  return f.v;
}
// B fragment from transposed weights layout (N x K row-major).
// Lane n (<16): col n, K {0..7, 16..23}; lane n+16: col n, K {8..15, 24..31}.
DEV v16bf load_frag_b(const unsigned short* wt, int ldk, int n0, int k0, int lane) {
  ABFrag f;
  const unsigned short* p = wt + (long)(n0 + (lane & 15)) * ldk + k0 + ((lane >> 4) << 3);
  f.q[0] = *reinterpret_cast<const uint4*>(p);
  f.q[1] = *reinterpret_cast<const uint4*>(p + 16);
  return f.v;
}

// generic __shared__ pointer -> 32-bit LDS byte offset (aperture keeps it in addr[31:0])
DEV unsigned lds_off32(const void* p) { return (unsigned)(size_t)p; }

// ---------------------------------------------------------------- WMMA GEMM
// out[M,N] = A[M,K](bf16) @ Wt[N,K](bf16)^T + bias (+residual) (relu) (rowmask->0)
// block = 256 threads = 8 waves (2x4), block tile 128x256, wave tile 64x64.
// K-step 32; A/B tiles double-buffered in LDS via global_load_async_to_lds_b128.
__global__ __launch_bounds__(256) void wmma_gemm_kernel(
    const unsigned short* __restrict__ A,
    const unsigned short* __restrict__ Wt,
    const float* __restrict__ bias,
    const float* __restrict__ residual,
    const unsigned char* __restrict__ rowmask,
    float* __restrict__ outF,
    unsigned short* __restrict__ outB,
    int M, int N, int K, int doRelu)
{
  // per buffer: A tile 128x32 (4096 shorts) at 0, B tile 256x32 (8192 shorts) at 4096
  __shared__ unsigned short lds[2][12288];

  const int tid  = threadIdx.x;
  const int lane = tid & 31;
  const int wv   = tid >> 5;
  const int mblk = blockIdx.x * 128;
  const int nblk = blockIdx.y * 256;
  const int mw   = (wv >> 2) * 64;    // wave tile row offset inside block tile
  const int nw   = (wv & 3) * 64;     // wave tile col offset inside block tile
  (void)M;

  // async-stage one K-slab (A: 512 b128 chunks, B: 1024 b128 chunks; 6 per thread)
  auto stage = [&](int buf, int k0) {
    unsigned short* As = &lds[buf][0];
    unsigned short* Bs = &lds[buf][4096];
#pragma unroll
    for (int c = tid; c < 512; c += 256) {                 // A tile
      const int row = c >> 2, ko = (c & 3) << 3;
      const unsigned long long ga =
          (unsigned long long)(A + (long)(mblk + row) * K + k0 + ko);
      const unsigned la = lds_off32(As + row * 32 + ko);
      asm volatile("global_load_async_to_lds_b128 %0, %1, off"
                   :: "v"(la), "v"(ga) : "memory");
    }
#pragma unroll
    for (int c = tid; c < 1024; c += 256) {                // B tile
      const int row = c >> 2, ko = (c & 3) << 3;
      const unsigned long long ga =
          (unsigned long long)(Wt + (long)(nblk + row) * K + k0 + ko);
      const unsigned la = lds_off32(Bs + row * 32 + ko);
      asm volatile("global_load_async_to_lds_b128 %0, %1, off"
                   :: "v"(la), "v"(ga) : "memory");
    }
  };

  v8f acc[4][4];
#pragma unroll
  for (int i = 0; i < 4; ++i)
#pragma unroll
    for (int j = 0; j < 4; ++j)
      acc[i][j] = (v8f){0.f, 0.f, 0.f, 0.f, 0.f, 0.f, 0.f, 0.f};

  stage(0, 0);
  int p = 0;
  for (int k0 = 0; k0 < K; k0 += 32, p ^= 1) {
    // every wave drains its own async queue, then barrier => all tiles visible
    asm volatile("s_wait_asynccnt 0x0" ::: "memory");
    __syncthreads();
    if (k0 + 32 < K) stage(p ^ 1, k0 + 32);   // prefetch next slab into other buffer

    const unsigned short* As = &lds[p][0];
    const unsigned short* Bs = &lds[p][4096];
    v16bf a[4], b[4];
#pragma unroll
    for (int i = 0; i < 4; ++i) a[i] = load_frag_a(As, 32, mw + 16 * i, 0, lane);
#pragma unroll
    for (int j = 0; j < 4; ++j) b[j] = load_frag_b(Bs, 32, nw + 16 * j, 0, lane);
#pragma unroll
    for (int i = 0; i < 4; ++i)
#pragma unroll
      for (int j = 0; j < 4; ++j)
        acc[i][j] = __builtin_amdgcn_wmma_f32_16x16x32_bf16(
            false, a[i], false, b[j], (short)0, acc[i][j], false, false);
    __syncthreads();   // all reads of lds[p] done before it is restaged
  }

  const int colb = lane & 15;
  const int rowb = (lane >> 4) * 8;
#pragma unroll
  for (int i = 0; i < 4; ++i) {
#pragma unroll
    for (int j = 0; j < 4; ++j) {
      const int col = nblk + nw + 16 * j + colb;
      const float bv = bias ? bias[col] : 0.f;
#pragma unroll
      for (int r = 0; r < 8; ++r) {
        const int row = mblk + mw + 16 * i + rowb + r;
        const long idx = (long)row * N + col;
        float v = acc[i][j][r] + bv;
        if (residual) v += residual[idx];
        if (doRelu)   v = fmaxf(v, 0.f);
        if (rowmask && rowmask[row]) v = 0.f;
        if (outF) outF[idx] = v;
        if (outB) outB[idx] = f2bf(v);
      }
    }
  }
}

// ---------------------------------------------------------------- conversions
__global__ __launch_bounds__(256) void to_bf16_kernel(const float* __restrict__ src,
                                                      unsigned short* __restrict__ dst, long n) {
  long i = (long)blockIdx.x * 256 + threadIdx.x;
  if (i < n) dst[i] = f2bf(src[i]);
}

__global__ __launch_bounds__(256) void transpose_bf16_kernel(const float* __restrict__ W,
                                                             unsigned short* __restrict__ Wt,
                                                             int K, int N) {
  long i = (long)blockIdx.x * 256 + threadIdx.x;
  if (i >= (long)K * N) return;
  int k = (int)(i / N), n = (int)(i % N);
  Wt[(long)n * K + k] = f2bf(W[i]);
}

__global__ __launch_bounds__(256) void copy_f32_kernel(const float* __restrict__ s,
                                                       float* __restrict__ d, int n) {
  int i = blockIdx.x * 256 + threadIdx.x;
  if (i < n) d[i] = s[i];
}

// ---------------------------------------------------------------- text features
__global__ __launch_bounds__(256) void text_feats_kernel(
    const float* __restrict__ text,
    const float* __restrict__ rw, const float* __restrict__ rb,
    const float* __restrict__ ofw, const float* __restrict__ ofb,
    const float* __restrict__ aww, const float* __restrict__ awb_,
    float* __restrict__ locOut, float* __restrict__ awOut)
{
  __shared__ float sq[256];
  __shared__ float sref[6];
  __shared__ float soff[192];
  __shared__ float sawr[96];
  __shared__ float saw[96];
  const int bl = blockIdx.x;          // b*32+lt
  const int tid = threadIdx.x;
  sq[tid] = text[(long)bl * 256 + tid];
  __syncthreads();
  for (int c = tid; c < 294; c += 256) {
    float acc;
    if (c < 6) {
      acc = rb[c];
      for (int k = 0; k < 256; ++k) acc += sq[k] * rw[k * 6 + c];
      sref[c] = 1.f / (1.f + expf(-acc));
    } else if (c < 198) {
      int cc = c - 6;
      acc = ofb[cc];
      for (int k = 0; k < 256; ++k) acc += sq[k] * ofw[k * 192 + cc];
      soff[cc] = acc;
    } else {
      int cc = c - 198;
      acc = awb_[cc];
      for (int k = 0; k < 256; ++k) acc += sq[k] * aww[k * 96 + cc];
      sawr[cc] = acc;
    }
  }
  __syncthreads();
  if (tid < 8) {                      // softmax over L*P=12 per head
    const int h = tid;
    float mx = -1e30f;
    for (int i = 0; i < 12; ++i) mx = fmaxf(mx, sawr[h * 12 + i]);
    float s = 0.f;
    for (int i = 0; i < 12; ++i) { float e = expf(sawr[h * 12 + i] - mx); saw[h * 12 + i] = e; s += e; }
    float inv = 1.f / s;
    for (int i = 0; i < 12; ++i) saw[h * 12 + i] *= inv;
  }
  __syncthreads();
  if (tid < 96) {
    const int h = tid / 12, r = tid % 12, l = r / 4, p = r % 4;
    const float nw_[3] = {96.f, 48.f, 24.f};
    const float nh_[3] = {72.f, 36.f, 18.f};
    float lx = sref[l * 2 + 0] + soff[((h * 3 + l) * 4 + p) * 2 + 0] / nw_[l];
    float ly = sref[l * 2 + 1] + soff[((h * 3 + l) * 4 + p) * 2 + 1] / nh_[l];
    long base = ((long)bl * 8 + h) * 12 + r;
    locOut[base * 2 + 0] = lx;
    locOut[base * 2 + 1] = ly;
    awOut[base] = saw[h * 12 + r];
  }
}

// ---------------------------------------------------------------- deform sampling
__global__ __launch_bounds__(32) void deform_sample_kernel(
    const float* __restrict__ value,   // (b*NV+n)*256 + h*32+dh
    const float* __restrict__ loc, const float* __restrict__ aw,
    float* __restrict__ ctx)
{
  const int lt = blockIdx.x, h = blockIdx.y, b = blockIdx.z, dh = threadIdx.x;
  const int Hl[3] = {72, 36, 18}, Wl[3] = {96, 48, 24}, ST[3] = {0, 6912, 8640};
  const long bl = (long)b * 32 + lt;
  const float* locp = loc + ((bl * 8 + h) * 12) * 2;
  const float* awp  = aw + (bl * 8 + h) * 12;
  float acc = 0.f;
  for (int l = 0; l < 3; ++l) {
    const int W_ = Wl[l], H_ = Hl[l], st = ST[l];
    for (int p = 0; p < 4; ++p) {
      const int r = l * 4 + p;
      const float x = locp[r * 2 + 0] * W_ - 0.5f;
      const float y = locp[r * 2 + 1] * H_ - 0.5f;
      const float x0 = floorf(x), y0 = floorf(y);
      const float wx = x - x0, wy = y - y0;
      float s = 0.f;
      for (int c = 0; c < 4; ++c) {
        const float xi = x0 + (float)(c & 1);
        const float yi = y0 + (float)(c >> 1);
        const float w = ((c & 1) ? wx : 1.f - wx) * ((c >> 1) ? wy : 1.f - wy);
        if (xi >= 0.f && xi < (float)W_ && yi >= 0.f && yi < (float)H_) {
          const int ix = (int)xi, iy = (int)yi;
          s += w * value[((long)b * NVv + st + (long)iy * W_ + ix) * 256 + h * 32 + dh];
        }
      }
      acc += awp[r] * s;
    }
  }
  ctx[bl * 256 + h * 32 + dh] = acc;
}

// ---------------------------------------------------------------- small GEMM (text path)
__global__ __launch_bounds__(256) void small_gemm_kernel(
    const float* __restrict__ A, const float* __restrict__ W,
    const float* __restrict__ bias, const float* __restrict__ residual,
    float* __restrict__ out, int K, int N, int doRelu)
{
  __shared__ float srow[1024];
  const int row = blockIdx.x, tid = threadIdx.x;
  for (int k = tid; k < K; k += 256) srow[k] = A[(long)row * K + k];
  __syncthreads();
  for (int col = tid; col < N; col += 256) {
    float acc = bias ? bias[col] : 0.f;
    for (int k = 0; k < K; ++k) acc += srow[k] * W[(long)k * N + col];
    if (residual) acc += residual[(long)row * N + col];
    if (doRelu) acc = fmaxf(acc, 0.f);
    out[(long)row * N + col] = acc;
  }
}

// ---------------------------------------------------------------- layer norms
__global__ __launch_bounds__(256) void ln_small_kernel(
    const float* __restrict__ in, const float* __restrict__ s,
    const float* __restrict__ b, float* __restrict__ out)
{
  __shared__ float r1[256];
  __shared__ float r2[256];
  const int row = blockIdx.x, tid = threadIdx.x;
  const float x = in[(long)row * 256 + tid];
  r1[tid] = x; r2[tid] = x * x;
  __syncthreads();
  for (int st = 128; st > 0; st >>= 1) {
    if (tid < st) { r1[tid] += r1[tid + st]; r2[tid] += r2[tid + st]; }
    __syncthreads();
  }
  const float m   = r1[0] * (1.f / 256.f);
  const float var = r2[0] * (1.f / 256.f) - m * m;
  const float inv = rsqrtf(var + 1e-5f);
  out[(long)row * 256 + tid] = (x - m) * inv * s[tid] + b[tid];
}

// big LN: 8 waves / block, one wave32 per row of 256, shuffle reduction
__global__ __launch_bounds__(256) void ln_big_kernel(
    const float* __restrict__ x, const float* __restrict__ sG,
    const float* __restrict__ bG, float* __restrict__ outF,
    unsigned short* __restrict__ outB, long rows)
{
  const int wid = threadIdx.x >> 5, lane = threadIdx.x & 31;
  const long row = (long)blockIdx.x * 8 + wid;
  if (row >= rows) return;
  const float* xp = x + row * 256;
  float v[8];
  float s = 0.f, s2 = 0.f;
#pragma unroll
  for (int k = 0; k < 8; ++k) {
    v[k] = xp[lane + 32 * k];
    s += v[k]; s2 += v[k] * v[k];
  }
#pragma unroll
  for (int o = 16; o >= 1; o >>= 1) {
    s  += __shfl_xor(s,  o, 32);
    s2 += __shfl_xor(s2, o, 32);
  }
  const float m   = s * (1.f / 256.f);
  const float var = s2 * (1.f / 256.f) - m * m;
  const float inv = rsqrtf(var + 1e-5f);
#pragma unroll
  for (int k = 0; k < 8; ++k) {
    const int c = lane + 32 * k;
    const float o = (v[k] - m) * inv * sG[c] + bG[c];
    outF[row * 256 + c] = o;
    if (outB) outB[row * 256 + c] = f2bf(o);
  }
}

// ---------------------------------------------------------------- MHA (32 keys)
__global__ __launch_bounds__(256) void mha_attn_kernel(
    const float* __restrict__ Q,  const float* __restrict__ Kb,
    const float* __restrict__ Vb, const unsigned char* __restrict__ tmask,
    float* __restrict__ ctx)
{
  const int b = blockIdx.z, h = blockIdx.y;
  __shared__ float sk[32 * 32];
  __shared__ float sv[32 * 32];
  __shared__ float sm[32];
  for (int i = threadIdx.x; i < 32 * 32; i += 256) {
    const int lk = i >> 5, dh = i & 31;
    sk[i] = Kb[((long)(b * 32 + lk)) * 256 + h * 32 + dh];
    sv[i] = Vb[((long)(b * 32 + lk)) * 256 + h * 32 + dh];
  }
  if (threadIdx.x < 32)
    sm[threadIdx.x] = tmask[b * 32 + threadIdx.x] ? -1e9f : 0.f;
  __syncthreads();

  const long q = (long)blockIdx.x * 256 + threadIdx.x;
  if (q >= NVv) return;
  const float* qp = Q + ((long)b * NVv + q) * 256 + h * 32;
  float qv[32];
#pragma unroll
  for (int d = 0; d < 32; ++d) qv[d] = qp[d];

  const float scale = 0.17677669529663687f;   // 1/sqrt(32)
  float sc[32];
  float mx = -1e30f;
  for (int lk = 0; lk < 32; ++lk) {
    float a = 0.f;
#pragma unroll
    for (int d = 0; d < 32; ++d) a += qv[d] * sk[lk * 32 + d];
    a = a * scale + sm[lk];
    sc[lk] = a;
    mx = fmaxf(mx, a);
  }
  float denom = 0.f;
  for (int lk = 0; lk < 32; ++lk) { sc[lk] = expf(sc[lk] - mx); denom += sc[lk]; }
  const float inv = 1.f / denom;
  float acc[32];
#pragma unroll
  for (int d = 0; d < 32; ++d) acc[d] = 0.f;
  for (int lk = 0; lk < 32; ++lk) {
    const float a = sc[lk] * inv;
#pragma unroll
    for (int d = 0; d < 32; ++d) acc[d] += a * sv[lk * 32 + d];
  }
  float* op = ctx + ((long)b * NVv + q) * 256 + h * 32;
#pragma unroll
  for (int d = 0; d < 32; ++d) op[d] = acc[d];
}

// ---------------------------------------------------------------- launcher
extern "C" void kernel_launch(void* const* d_in, const int* in_sizes, int n_in,
                              void* d_out, int out_size, void* d_ws, size_t ws_size,
                              hipStream_t stream) {
  (void)in_sizes; (void)n_in; (void)out_size; (void)ws_size;
  const float* vis_tokens  = (const float*)d_in[0];
  const float* text_tokens = (const float*)d_in[1];
  const float* vis_value   = (const float*)d_in[2];
  const float* ref_w = (const float*)d_in[3];
  const float* ref_b = (const float*)d_in[4];
  const float* off_w = (const float*)d_in[5];
  const float* off_b = (const float*)d_in[6];
  const float* aw_w  = (const float*)d_in[7];
  const float* aw_b  = (const float*)d_in[8];
  const float* vproj_w = (const float*)d_in[9];
  const float* vproj_b = (const float*)d_in[10];
  const float* oproj_w = (const float*)d_in[11];
  const float* oproj_b = (const float*)d_in[12];
  const float* lvi_out_w = (const float*)d_in[13];
  const float* lvi_out_b = (const float*)d_in[14];
  const float* lvi_n1_s = (const float*)d_in[15];
  const float* lvi_n1_b = (const float*)d_in[16];
  const float* lvi_f1_w = (const float*)d_in[17];
  const float* lvi_f1_b = (const float*)d_in[18];
  const float* lvi_f2_w = (const float*)d_in[19];
  const float* lvi_f2_b = (const float*)d_in[20];
  const float* lvi_n2_s = (const float*)d_in[21];
  const float* lvi_n2_b = (const float*)d_in[22];
  const float* mha_qw = (const float*)d_in[23];
  const float* mha_qb = (const float*)d_in[24];
  const float* mha_kw = (const float*)d_in[25];
  const float* mha_kb = (const float*)d_in[26];
  const float* mha_vw = (const float*)d_in[27];
  const float* mha_vb = (const float*)d_in[28];
  const float* mha_ow = (const float*)d_in[29];
  const float* mha_ob = (const float*)d_in[30];
  const float* vli_out_w = (const float*)d_in[31];
  const float* vli_out_b = (const float*)d_in[32];
  const float* vli_n1_s = (const float*)d_in[33];
  const float* vli_n1_b = (const float*)d_in[34];
  const float* vli_f1_w = (const float*)d_in[35];
  const float* vli_f1_b = (const float*)d_in[36];
  const float* vli_f2_w = (const float*)d_in[37];
  const float* vli_f2_b = (const float*)d_in[38];
  const float* vli_n2_s = (const float*)d_in[39];
  const float* vli_n2_b = (const float*)d_in[40];
  const unsigned char* text_mask = (const unsigned char*)d_in[43];
  const unsigned char* vis_mask  = (const unsigned char*)d_in[44];

  // -------- workspace carve-up (all chunks are multiples of 256 B)
  char* w = (char*)d_ws;
  size_t o = 0;
  float* F1 = (float*)(w + o);            o += (size_t)BN * 256 * 4;      // 74.3 MB
  float* F2 = (float*)(w + o);            o += (size_t)BN * 256 * 4;      // 74.3 MB
  unsigned short* A16 = (unsigned short*)(w + o); o += (size_t)BN * 256 * 2;     // 37.2 MB
  unsigned short* X16 = (unsigned short*)(w + o); o += (size_t)CHUNK * 1024 * 2; // 49.5 MB
  unsigned short* wt_vproj  = (unsigned short*)(w + o); o += 256 * 256 * 2;
  unsigned short* wt_qw     = (unsigned short*)(w + o); o += 256 * 256 * 2;
  unsigned short* wt_ow     = (unsigned short*)(w + o); o += 256 * 256 * 2;
  unsigned short* wt_vliout = (unsigned short*)(w + o); o += 256 * 256 * 2;
  unsigned short* wt_f1     = (unsigned short*)(w + o); o += 256 * 1024 * 2;
  unsigned short* wt_f2     = (unsigned short*)(w + o); o += 1024 * 256 * 2;
  float* locb    = (float*)(w + o); o += (size_t)Bsz * Lt * Hh * 12 * 2 * 4;
  float* awb     = (float*)(w + o); o += (size_t)Bsz * Lt * Hh * 12 * 4;
  float* ctxb    = (float*)(w + o); o += (size_t)Bsz * Lt * 256 * 4;
  float* tmp1    = (float*)(w + o); o += (size_t)Bsz * Lt * 1024 * 4;
  float* tmp2    = (float*)(w + o); o += (size_t)Bsz * Lt * 256 * 4;
  float* tpre    = (float*)(w + o); o += (size_t)Bsz * Lt * 256 * 4;
  float* tbuf    = (float*)(w + o); o += (size_t)Bsz * Lt * 256 * 4;
  float* textout = (float*)(w + o); o += (size_t)Bsz * Lt * 256 * 4;
  float* kbuf    = (float*)(w + o); o += (size_t)Bsz * Lt * 256 * 4;
  float* vbuf    = (float*)(w + o); o += (size_t)Bsz * Lt * 256 * 4;

  const long nBig = BN * 256;                          // 18 579 456
  const int  gConv = (int)((nBig + 255) / 256);
  const dim3 gemmGrid256((int)(BN / 128), 1);          // 567 x 1
  const dim3 gemmGridC1024((int)(CHUNK / 128), 4);     // 189 x 4
  const dim3 gemmGridC256((int)(CHUNK / 128), 1);      // 189 x 1

  // -------- weight prep (bf16, transposed to N x K)
  transpose_bf16_kernel<<<256,  256, 0, stream>>>(vproj_w,   wt_vproj,  256, 256);
  transpose_bf16_kernel<<<256,  256, 0, stream>>>(mha_qw,    wt_qw,     256, 256);
  transpose_bf16_kernel<<<256,  256, 0, stream>>>(mha_ow,    wt_ow,     256, 256);
  transpose_bf16_kernel<<<256,  256, 0, stream>>>(vli_out_w, wt_vliout, 256, 256);
  transpose_bf16_kernel<<<1024, 256, 0, stream>>>(vli_f1_w,  wt_f1,     256, 1024);
  transpose_bf16_kernel<<<1024, 256, 0, stream>>>(vli_f2_w,  wt_f2,     1024, 256);

  // -------- value projection (big WMMA GEMM), masked rows -> 0
  to_bf16_kernel<<<gConv, 256, 0, stream>>>(vis_value, A16, nBig);
  wmma_gemm_kernel<<<gemmGrid256, 256, 0, stream>>>(A16, wt_vproj, vproj_b, nullptr,
      vis_mask, F1, nullptr, (int)BN, 256, 256, 0);

  // -------- text path: sampling locations / attention weights, deform sample
  text_feats_kernel<<<Bsz * Lt, 256, 0, stream>>>(text_tokens, ref_w, ref_b,
      off_w, off_b, aw_w, aw_b, locb, awb);
  deform_sample_kernel<<<dim3(Lt, Hh, Bsz), 32, 0, stream>>>(F1, locb, awb, ctxb);

  // ctx @ oproj ; + lvi_out projection + residual ; LN1 ; FFN ; LN2
  small_gemm_kernel<<<Bsz * Lt, 256, 0, stream>>>(ctxb, oproj_w, oproj_b, nullptr, tmp2, 256, 256, 0);
  small_gemm_kernel<<<Bsz * Lt, 256, 0, stream>>>(tmp2, lvi_out_w, lvi_out_b, text_tokens, tpre, 256, 256, 0);
  ln_small_kernel<<<Bsz * Lt, 256, 0, stream>>>(tpre, lvi_n1_s, lvi_n1_b, tbuf);
  small_gemm_kernel<<<Bsz * Lt, 256, 0, stream>>>(tbuf, lvi_f1_w, lvi_f1_b, nullptr, tmp1, 256, 1024, 1);
  small_gemm_kernel<<<Bsz * Lt, 256, 0, stream>>>(tmp1, lvi_f2_w, lvi_f2_b, tbuf, tmp2, 1024, 256, 0);
  ln_small_kernel<<<Bsz * Lt, 256, 0, stream>>>(tmp2, lvi_n2_s, lvi_n2_b, textout);
  copy_f32_kernel<<<(Bsz * Lt * 256) / 256, 256, 0, stream>>>(
      textout, (float*)d_out + (size_t)BN * 256, Bsz * Lt * 256);

  // -------- MHA: K/V from text_out (small), Q from vis_tokens (big WMMA)
  small_gemm_kernel<<<Bsz * Lt, 256, 0, stream>>>(textout, mha_kw, mha_kb, nullptr, kbuf, 256, 256, 0);
  small_gemm_kernel<<<Bsz * Lt, 256, 0, stream>>>(textout, mha_vw, mha_vb, nullptr, vbuf, 256, 256, 0);

  to_bf16_kernel<<<gConv, 256, 0, stream>>>(vis_tokens, A16, nBig);
  wmma_gemm_kernel<<<gemmGrid256, 256, 0, stream>>>(A16, wt_qw, mha_qb, nullptr,
      nullptr, F2, nullptr, (int)BN, 256, 256, 0);

  mha_attn_kernel<<<dim3((int)((NVv + 255) / 256), Hh, Bsz), 256, 0, stream>>>(
      F2, kbuf, vbuf, text_mask, F1);

  // ctx_v @ mha_ow (bf16 out) ; @ vli_out_w + vis_tokens residual (f32 out)
  to_bf16_kernel<<<gConv, 256, 0, stream>>>(F1, A16, nBig);
  wmma_gemm_kernel<<<gemmGrid256, 256, 0, stream>>>(A16, wt_ow, mha_ob, nullptr,
      nullptr, nullptr, X16, (int)BN, 256, 256, 0);
  wmma_gemm_kernel<<<gemmGrid256, 256, 0, stream>>>(X16, wt_vliout, vli_out_b, vis_tokens,
      nullptr, F2, nullptr, (int)BN, 256, 256, 0);

  // LN1 -> v (fp32 in F1 for residual, bf16 in A16 for FFN)
  ln_big_kernel<<<(int)(BN / 8), 256, 0, stream>>>(F2, vli_n1_s, vli_n1_b, F1, A16, BN);

  // FFN in 3 row chunks, hidden (bf16) reuses X16
  for (int c = 0; c < 3; ++c) {
    const long r0 = (long)c * CHUNK;
    wmma_gemm_kernel<<<gemmGridC1024, 256, 0, stream>>>(A16 + r0 * 256, wt_f1, vli_f1_b,
        nullptr, nullptr, nullptr, X16, (int)CHUNK, 1024, 256, 1);
    wmma_gemm_kernel<<<gemmGridC256, 256, 0, stream>>>(X16, wt_f2, vli_f2_b,
        F1 + r0 * 256, nullptr, F2 + r0 * 256, nullptr, (int)CHUNK, 256, 1024, 0);
  }

  // LN2 -> vis_out (first section of d_out)
  ln_big_kernel<<<(int)(BN / 8), 256, 0, stream>>>(F2, vli_n2_s, vli_n2_b,
      (float*)d_out, nullptr, BN);
}
